// PHYNN2DOF_82446192214227
// MI455X (gfx1250) — matrix operand order repaired and checked by reading, not compiled, not run on hardware
//
#include <hip/hip_runtime.h>

typedef __attribute__((ext_vector_type(16))) _Float16 v16h;
typedef __attribute__((ext_vector_type(8)))  _Float16 v8h;
typedef __attribute__((ext_vector_type(8)))  float    v8f;

#define TPB 256
#define WAVES_PER_BLOCK 8

__device__ __forceinline__ float fast_tanh(float x) {
#if __has_builtin(__builtin_amdgcn_tanhf)
    return __builtin_amdgcn_tanhf(x);
#elif __has_builtin(__builtin_amdgcn_exp2f) && __has_builtin(__builtin_amdgcn_rcpf)
    // tanh(x) = 1 - 2/(exp(2x)+1);  exp(2x) = exp2(x * 2*log2(e))
    float e = __builtin_amdgcn_exp2f(x * 2.88539008177792681472f);
    return 1.0f - 2.0f * __builtin_amdgcn_rcpf(e + 1.0f);
#else
    float e = __expf(2.0f * x);
    return 1.0f - 2.0f / (e + 1.0f);
#endif
}

__global__ __launch_bounds__(TPB) void phynn2dof_wmma_kernel(
    const float* __restrict__ q,  const float* __restrict__ qd, const float* __restrict__ u,
    const float* __restrict__ Wm1, const float* __restrict__ bm1,
    const float* __restrict__ Wm2, const float* __restrict__ bm2,
    const float* __restrict__ Wc1, const float* __restrict__ bc1,
    const float* __restrict__ Wc2, const float* __restrict__ bc2,
    const float* __restrict__ Wg1, const float* __restrict__ bg1,
    const float* __restrict__ Wg2, const float* __restrict__ bg2,
    const float* __restrict__ Wt1, const float* __restrict__ bt1,
    const float* __restrict__ Wt2, const float* __restrict__ bt2,
    float* __restrict__ out, int Bn)
{
    // Layer-1 A-fragments: 32 tiles x 32 lanes x 8 halfs (elements 8..15 are zero -> in regs)
    __shared__ __align__(16) _Float16 sW1[32][32][8];    // 16 KB
    // Layer-2 A-fragments, K ordering sigma-permuted: 16 chunks x 32 lanes x 16 halfs
    __shared__ __align__(32) _Float16 sA2[16][32][16];   // 16 KB
    __shared__ float sB1[512];                           // combined layer-1 bias
    __shared__ float sB2[16];                            // combined layer-2 bias

    const int t = threadIdx.x;

    // ---- stage combined layer-1 weights (A operand layout: lanes 0-15 hold K 0..7) ----
    for (int idx = t; idx < 32*32*8; idx += TPB) {
        int n  = idx >> 8;
        int l  = (idx >> 3) & 31;
        int kk = idx & 7;
        float v = 0.0f;
        if (l < 16 && kk < 6) {
            int h = 16*n + l;                 // combined hidden index
            if      (h < 128) { if (kk < 2) v = Wm1[h*2 + kk]; }
            else if (h < 256) { if (kk < 4) v = Wc1[(h-128)*4 + kk]; }
            else if (h < 384) { if (kk < 2) v = Wg1[(h-256)*2 + kk]; }
            else              {             v = Wt1[(h-384)*6 + kk]; }
        }
        sW1[n][l][kk] = (_Float16)v;
    }
    // ---- stage combined layer-2 weights with sigma(K) so layer-1 D == layer-2 B fragment ----
    for (int idx = t; idx < 16*32*16; idx += TPB) {
        int j  = idx >> 9;
        int l  = (idx >> 4) & 31;
        int m  = idx & 15;
        int kb = (l < 16) ? 0 : 8;
        int K  = (m < 8) ? (kb + m) : (16 + kb + (m - 8));          // A-fragment K coord
        int Ks = K;                                                 // identity: D rows already in table order
        int k  = 32*j + Ks;                                         // combined hidden index
        int r  = l & 15;                                            // output row
        float v = 0.0f;
        if      (r < 4)  { if (k < 128)             v = Wm2[r*128 + k];             }
        else if (r < 8)  { if (k >= 128 && k < 256) v = Wc2[(r-4)*128 + (k-128)];   }
        else if (r < 10) { if (k >= 256 && k < 384) v = Wg2[(r-8)*128 + (k-256)];   }
        else if (r < 12) { if (k >= 384)            v = Wt2[(r-10)*128 + (k-384)];  }
        sA2[j][l][m] = (_Float16)v;
    }
    // ---- stage combined biases ----
    for (int idx = t; idx < 512; idx += TPB) {
        float v;
        if      (idx < 128) v = bm1[idx];
        else if (idx < 256) v = bc1[idx-128];
        else if (idx < 384) v = bg1[idx-256];
        else                v = bt1[idx-384];
        sB1[idx] = v;
    }
    if (t < 16) {
        float v = 0.0f;
        if      (t < 4)  v = bm2[t];
        else if (t < 8)  v = bc2[t-4];
        else if (t < 10) v = bg2[t-8];
        else if (t < 12) v = bt2[t-10];
        sB2[t] = v;
    }
    __syncthreads();

    const int lane = t & 31;
    const int s    = lane & 15;          // sample column within tile
    const int hi   = lane >> 4;          // half-wave
    const int NT   = Bn >> 4;            // 16 samples per tile
    const int gw0  = blockIdx.x * WAVES_PER_BLOCK + (t >> 5);
    const int wstride = gridDim.x * WAVES_PER_BLOCK;

    for (int tile = gw0; tile < NT; tile += wstride) {
        const int gs = tile * 16 + s;
        const float2 qv  = ((const float2*)q )[gs];
        const float2 qdv = ((const float2*)qd)[gs];
        const float2 uv  = ((const float2*)u )[gs];

        // B operand for layer 1: x^T (K down, samples across lanes 0-15; K>=6 zero)
        v16h xb = {};
        if (hi == 0) {
            xb[0] = (_Float16)qv.x;  xb[1] = (_Float16)qv.y;
            xb[2] = (_Float16)qdv.x; xb[3] = (_Float16)qdv.y;
            xb[4] = (_Float16)uv.x;  xb[5] = (_Float16)uv.y;
        }

        // layer-2 accumulator seeded with layer-2 bias (rows r = i + 8*hi)
        v8f acc2;
        #pragma unroll
        for (int i = 0; i < 8; ++i) acc2[i] = sB2[8*hi + i];

        #pragma unroll 4
        for (int j = 0; j < 16; ++j) {
            // ---- layer 1: tiles n = 2j, 2j+1 ; C seeded with bias ----
            v8f c0, c1;
            const int bb = 32*j + 8*hi;
            #pragma unroll
            for (int i = 0; i < 8; ++i) { c0[i] = sB1[bb + i]; c1[i] = sB1[bb + 16 + i]; }

            v8h w0 = *(const v8h*)&sW1[2*j    ][lane][0];
            v8h w1 = *(const v8h*)&sW1[2*j + 1][lane][0];
            v16h a0 = {}, a1 = {};
            #pragma unroll
            for (int i = 0; i < 8; ++i) { a0[i] = w0[i]; a1[i] = w1[i]; }

            c0 = __builtin_amdgcn_wmma_f32_16x16x32_f16(false, a0, false, xb, (short)0, c0, false, false);
            c1 = __builtin_amdgcn_wmma_f32_16x16x32_f16(false, a1, false, xb, (short)0, c1, false, false);

            // ---- activation: D(f32) -> B(f16) fragment directly (sigma baked into sA2) ----
            v16h hb;
            #pragma unroll
            for (int i = 0; i < 8; ++i) {
                hb[i]     = (_Float16)fast_tanh(c0[i]);
                hb[8 + i] = (_Float16)fast_tanh(c1[i]);
            }

            // ---- layer 2 accumulate ----
            v16h a2 = *(const v16h*)&sA2[j][lane][0];
            acc2 = __builtin_amdgcn_wmma_f32_16x16x32_f16(false, a2, false, hb, (short)0, acc2, false, false);
        }

        // rows 8..11 (g0,g1,t0,t1) live in the upper half-wave's acc2[0..3]
        const float g0 = __shfl(acc2[0], s + 16, 32);
        const float g1 = __shfl(acc2[1], s + 16, 32);
        const float t0 = __shfl(acc2[2], s + 16, 32);
        const float t1 = __shfl(acc2[3], s + 16, 32);

        if (hi == 0) {
            // rows 0..3 = L (row-major 2x2, tril applied), rows 4..7 = C
            const float L00 = acc2[0], L10 = acc2[2], L11 = acc2[3];
            const float C00 = acc2[4], C01 = acc2[5], C10 = acc2[6], C11 = acc2[7];
            const float M00 = L00*L00 + 1e-3f;
            const float M01 = L00*L10;
            const float M11 = L10*L10 + L11*L11 + 1e-3f;
            const float r0  = uv.x - (C00*qdv.x + C01*qdv.y) - g0 - t0;
            const float r1  = uv.y - (C10*qdv.x + C11*qdv.y) - g1 - t1;
            const float inv = 1.0f / (M00*M11 - M01*M01);
            const float qdd0 = (M11*r0 - M01*r1) * inv;
            const float qdd1 = (M00*r1 - M01*r0) * inv;

            const long Bl = (long)Bn;
            ((float2*)(out          ))[gs] = make_float2(qdd0, qdd1);
            float4 Mv; Mv.x = M00; Mv.y = M01; Mv.z = M01; Mv.w = M11;
            ((float4*)(out + 2L*Bl ))[gs] = Mv;
            float4 Cv; Cv.x = C00; Cv.y = C01; Cv.z = C10; Cv.w = C11;
            ((float4*)(out + 6L*Bl ))[gs] = Cv;
            ((float2*)(out + 10L*Bl))[gs] = make_float2(g0, g1);
            ((float2*)(out + 12L*Bl))[gs] = make_float2(t0, t1);
        }
    }
}

extern "C" void kernel_launch(void* const* d_in, const int* in_sizes, int n_in,
                              void* d_out, int out_size, void* d_ws, size_t ws_size,
                              hipStream_t stream) {
    const float* q   = (const float*)d_in[0];
    const float* qd  = (const float*)d_in[1];
    const float* u   = (const float*)d_in[2];
    const float* Wm1 = (const float*)d_in[3];
    const float* bm1 = (const float*)d_in[4];
    const float* Wm2 = (const float*)d_in[5];
    const float* bm2 = (const float*)d_in[6];
    const float* Wc1 = (const float*)d_in[7];
    const float* bc1 = (const float*)d_in[8];
    const float* Wc2 = (const float*)d_in[9];
    const float* bc2 = (const float*)d_in[10];
    const float* Wg1 = (const float*)d_in[11];
    const float* bg1 = (const float*)d_in[12];
    const float* Wg2 = (const float*)d_in[13];
    const float* bg2 = (const float*)d_in[14];
    const float* Wt1 = (const float*)d_in[15];
    const float* bt1 = (const float*)d_in[16];
    const float* Wt2 = (const float*)d_in[17];
    const float* bt2 = (const float*)d_in[18];

    const int Bn = in_sizes[0] / 2;   // q is [B,2]
    const int blocks = 2048;          // grid-stride over Bn/16 tiles

    phynn2dof_wmma_kernel<<<blocks, TPB, 0, stream>>>(
        q, qd, u,
        Wm1, bm1, Wm2, bm2,
        Wc1, bc1, Wc2, bc2,
        Wg1, bg1, Wg2, bg2,
        Wt1, bt1, Wt2, bt2,
        (float*)d_out, Bn);
}
